// BEVSampling_7533372637355
// MI455X (gfx1250) — matrix operand order repaired and checked by reading, not compiled, not run on hardware
//
#include <hip/hip_runtime.h>
#include <hip/hip_bf16.h>

typedef __attribute__((ext_vector_type(16))) _Float16 v16h;
typedef __attribute__((ext_vector_type(8)))  _Float16 v8h;
typedef __attribute__((ext_vector_type(8)))  float    v8f;

#define NUM_ROWS   80000      // P(8) * Q(10000)
#define ROW_STRIDE 80000      // out layout [C][P][H][W] -> channel stride
#define HSTR       264        // padded LDS stride (halves) for hidden tile

// ---------------------------------------------------------------------------
// Kernel A: w2 [256,128] f32 -> w2t [128,256] f16 (B-operand friendly layout)
// ---------------------------------------------------------------------------
__global__ __launch_bounds__(256) void bev_w2t_kernel(const float* __restrict__ w2,
                                                      _Float16* __restrict__ w2t) {
    int e = blockIdx.x * 256 + threadIdx.x;      // 32768 elements
    int n = e >> 8;                              // output channel 0..127
    int k = e & 255;                             // hidden index 0..255
    w2t[e] = (_Float16)w2[k * 128 + n];
}

// ---------------------------------------------------------------------------
// Kernel B: NCHW [6,128,H,W] -> NHWC [6,H,W,128] repack (per level)
// ---------------------------------------------------------------------------
__global__ __launch_bounds__(256) void bev_nhwc_kernel(const float* __restrict__ src,
                                                       float* __restrict__ dst,
                                                       int H, int W, int total) {
    int e = blockIdx.x * 256 + threadIdx.x;
    if (e >= total) return;
    int c = e & 127;
    int s = e >> 7;
    int x = s % W;
    int y = (s / W) % H;
    int n = s / (W * H);
    dst[e] = src[((n * 128 + c) * H + y) * W + x];
}

// ---------------------------------------------------------------------------
// Kernel C: positional MLP via WMMA, specialized at compile time on the
// B-operand source so the K loop is exactly:
//   2x ds_load_b128 (A) + 2x global_load_b128 (B) + v_wmma_f32_16x16x32_f16
// Block: 256 threads = 8 waves; each wave owns one 16-channel output tile.
// ---------------------------------------------------------------------------
template <bool USE_W2T>
__global__ __launch_bounds__(256) void bev_mlp_kernel(const float* __restrict__ ref,
                                                      const float* __restrict__ w1,
                                                      const float* __restrict__ b1,
                                                      const float* __restrict__ w2,
                                                      const _Float16* __restrict__ w2t,
                                                      const float* __restrict__ b2,
                                                      float* __restrict__ out) {
    __shared__ __align__(16) _Float16 hid[16 * HSTR];
    __shared__ float stage[8 * 256];
    __shared__ float sref[48];

    int t = threadIdx.x;
    int rowBase = blockIdx.x * 16;

    if (t < 48) sref[t] = ref[rowBase * 3 + t];
    __syncthreads();

    // Layer 1: each thread computes hid[i][t] for i = 0..15 (K=3 GEMM + relu)
    float wa = w1[t], wb = w1[256 + t], wc = w1[512 + t], bb1 = b1[t];
#pragma unroll
    for (int i = 0; i < 16; ++i) {
        float h = fmaf(sref[i * 3 + 0], wa,
                  fmaf(sref[i * 3 + 1], wb,
                  fmaf(sref[i * 3 + 2], wc, bb1)));
        hid[i * HSTR + t] = (_Float16)fmaxf(h, 0.0f);
    }
    __syncthreads();

    int lane = t & 31;
    int wv   = t >> 5;          // wave id -> output channel tile
    int half = lane >> 4;       // 0: lanes 0-15, 1: lanes 16-31
    int m    = lane & 15;       // row-in-tile (A) / col-in-tile (B,C)
    int n    = wv * 16 + m;

    float cb = b2[n];
    v8f c = {cb, cb, cb, cb, cb, cb, cb, cb};

    // Layer 2: D = A(16x256 f16, LDS) x B(256x16 f16, global) + C
#pragma unroll
    for (int k0 = 0; k0 < 256; k0 += 32) {
        // A contiguous K order (matches B): lane<16 holds K k0..k0+15,
        // lane>=16 holds K k0+16..k0+31
        const v8h alo = *(const v8h*)&hid[m * HSTR + k0 + half * 16];
        const v8h ahi = *(const v8h*)&hid[m * HSTR + k0 + half * 16 + 8];
        v16h a = __builtin_shufflevector(alo, ahi,
                                         0, 1, 2, 3, 4, 5, 6, 7,
                                         8, 9, 10, 11, 12, 13, 14, 15);
        // B contiguous K order (matches A): N = lane&15, lanes<16 K=k0..k0+15,
        // lanes>=16 K=k0+16..k0+31. w2t is [n][k] contiguous -> one 32B load.
        v16h b;
        if constexpr (USE_W2T) {
            b = *(const v16h*)&w2t[n * 256 + k0 + half * 16];
        } else {
#pragma unroll
            for (int j = 0; j < 16; ++j)
                b[j] = (_Float16)w2[(k0 + half * 16 + j) * 128 + n];
        }
        c = __builtin_amdgcn_wmma_f32_16x16x32_f16(
                /*neg_a=*/false, a, /*neg_b=*/false, b,
                /*c_mod=*/(short)0, c, /*reuse_a=*/false, /*reuse_b=*/false);
    }

    // Stage D tile channel-major so the global write is row-coalesced.
    // c[r] holds element (M = r + half*8, N = m).
#pragma unroll
    for (int r = 0; r < 8; ++r)
        stage[wv * 256 + m * 16 + (r + half * 8)] = c[r];
    __syncthreads();

    // Write pos to out[ch * 80000 + rowBase + rr] in 16-wide coalesced runs.
#pragma unroll
    for (int i = 0; i < 8; ++i) {
        int ch = i * 16 + (t >> 4);
        int rr = t & 15;
        out[ch * ROW_STRIDE + rowBase + rr] = stage[i * 256 + (t >> 4) * 16 + rr];
    }
}

// ---------------------------------------------------------------------------
// Kernel D: projection + 4-level bilinear sampling, out += sum_cams(mean_lvls)
// One lane per (p,q) row; channels looped in groups of 4 (float4 NHWC gathers).
// Specialized at compile time on the feature layout.
// ---------------------------------------------------------------------------
template <bool USE_NHWC>
__global__ __launch_bounds__(256) void bev_samp_kernel(const float* __restrict__ f0,
                                                       const float* __restrict__ f1,
                                                       const float* __restrict__ f2,
                                                       const float* __restrict__ f3,
                                                       const float* __restrict__ ref,
                                                       const float* __restrict__ l2i,
                                                       const float* __restrict__ nhwc,
                                                       float* __restrict__ out) {
    __shared__ float sm[96];                 // 6 cameras x 4x4 lidar2img
    int t = threadIdx.x;
    if (t < 96) sm[t] = l2i[t];
    __syncthreads();

    int row = blockIdx.x * 256 + t;
    bool active = row < NUM_ROWS;
    int r = active ? row : 0;

    // denormalize to pc_range [-50,-50,-4, 50,50,4]
    float X = ref[r * 3 + 0] * 100.0f - 50.0f;
    float Y = ref[r * 3 + 1] * 100.0f - 50.0f;
    float Z = ref[r * 3 + 2] * 8.0f - 4.0f;

    float us[6], vs[6];
    bool  val[6];
#pragma unroll
    for (int cam = 0; cam < 6; ++cam) {
        const float* M = &sm[cam * 16];
        float cx = fmaf(M[0], X, fmaf(M[1], Y, fmaf(M[2],  Z, M[3])));
        float cy = fmaf(M[4], X, fmaf(M[5], Y, fmaf(M[6],  Z, M[7])));
        float cz = fmaf(M[8], X, fmaf(M[9], Y, fmaf(M[10], Z, M[11])));
        float zs = fmaxf(cz, 1e-6f);
        float u = cx / (zs * 704.0f);
        float v = cy / (zs * 256.0f);
        us[cam] = u;
        vs[cam] = v;
        val[cam] = (cz > 1e-6f) & (u > 0.0f) & (u < 1.0f) & (v > 0.0f) & (v < 1.0f);
    }

    const int Hs[4] = {32, 16, 8, 4};
    const int Ws[4] = {88, 44, 22, 11};
    const int loff[4] = {0, 2162688, 2703360, 2838528};   // NHWC level bases (floats)
    const float* fps[4] = {f0, f1, f2, f3};

#pragma unroll 1
    for (int cg = 0; cg < 32; ++cg) {
        float a0 = 0.0f, a1 = 0.0f, a2 = 0.0f, a3 = 0.0f;
#pragma unroll
        for (int cam = 0; cam < 6; ++cam) {
            if (!val[cam]) continue;
#pragma unroll
            for (int l = 0; l < 4; ++l) {
                int H = Hs[l], W = Ws[l];
                float xf = us[cam] * (float)W - 0.5f;
                float yf = vs[cam] * (float)H - 0.5f;
                float x0f = floorf(xf), y0f = floorf(yf);
                float wx = xf - x0f, wy = yf - y0f;
                int x0 = (int)x0f, y0 = (int)y0f;
#pragma unroll
                for (int ty = 0; ty < 2; ++ty) {
#pragma unroll
                    for (int tx = 0; tx < 2; ++tx) {
                        int xi = x0 + tx, yi = y0 + ty;
                        if (xi < 0 || xi >= W || yi < 0 || yi >= H) continue;
                        float wt = (tx ? wx : 1.0f - wx) * (ty ? wy : 1.0f - wy);
                        if constexpr (USE_NHWC) {
                            const float4 v4 = *(const float4*)(nhwc + loff[l]
                                + ((cam * H + yi) * W + xi) * 128 + cg * 4);
                            a0 = fmaf(wt, v4.x, a0);
                            a1 = fmaf(wt, v4.y, a1);
                            a2 = fmaf(wt, v4.z, a2);
                            a3 = fmaf(wt, v4.w, a3);
                        } else {
                            const float* fp = fps[l];
                            int hw = H * W;
                            int base = ((cam * 128 + cg * 4) * H + yi) * W + xi;
                            a0 = fmaf(wt, fp[base],          a0);
                            a1 = fmaf(wt, fp[base + hw],     a1);
                            a2 = fmaf(wt, fp[base + 2 * hw], a2);
                            a3 = fmaf(wt, fp[base + 3 * hw], a3);
                        }
                    }
                }
            }
        }
        if (active) {
            int i0 = (cg * 4) * ROW_STRIDE + row;    // out already holds pos
            out[i0]                  += 0.25f * a0;
            out[i0 + ROW_STRIDE]     += 0.25f * a1;
            out[i0 + 2 * ROW_STRIDE] += 0.25f * a2;
            out[i0 + 3 * ROW_STRIDE] += 0.25f * a3;
        }
    }
}

// ---------------------------------------------------------------------------
extern "C" void kernel_launch(void* const* d_in, const int* in_sizes, int n_in,
                              void* d_out, int out_size, void* d_ws, size_t ws_size,
                              hipStream_t stream) {
    (void)in_sizes; (void)n_in; (void)out_size;
    const float* f0  = (const float*)d_in[0];
    const float* f1  = (const float*)d_in[1];
    const float* f2  = (const float*)d_in[2];
    const float* f3  = (const float*)d_in[3];
    const float* ref = (const float*)d_in[4];   // [1,8,100,100,3]
    const float* l2i = (const float*)d_in[5];   // [1,6,4,4]
    const float* w1  = (const float*)d_in[6];   // [3,256]
    const float* b1  = (const float*)d_in[7];   // [256]
    const float* w2  = (const float*)d_in[8];   // [256,128]
    const float* b2  = (const float*)d_in[9];   // [128]
    float* out = (float*)d_out;

    const size_t W2T_BYTES  = 32768 * sizeof(_Float16);          // 64 KB
    const size_t NHWC_FLTS  = 2872320;                           // all 4 levels
    const size_t NEED_NHWC  = W2T_BYTES + NHWC_FLTS * sizeof(float);
    bool useW2T  = (ws_size >= W2T_BYTES);
    bool useNHWC = (ws_size >= NEED_NHWC);

    _Float16* w2t = (_Float16*)d_ws;
    float* nhwc = (float*)((char*)d_ws + W2T_BYTES);

    if (useW2T)
        bev_w2t_kernel<<<128, 256, 0, stream>>>(w2, w2t);

    if (useNHWC) {
        const int Hs[4] = {32, 16, 8, 4};
        const int Ws[4] = {88, 44, 22, 11};
        const int off[4] = {0, 2162688, 2703360, 2838528};
        const float* fs[4] = {f0, f1, f2, f3};
        for (int l = 0; l < 4; ++l) {
            int total = 6 * Hs[l] * Ws[l] * 128;
            int blocks = (total + 255) / 256;
            bev_nhwc_kernel<<<blocks, 256, 0, stream>>>(fs[l], nhwc + off[l],
                                                        Hs[l], Ws[l], total);
        }
    }

    // pos -> out (WMMA GEMM), then sampling accumulates on top (stream-ordered)
    if (useW2T)
        bev_mlp_kernel<true><<<NUM_ROWS / 16, 256, 0, stream>>>(ref, w1, b1, w2,
                                                                w2t, b2, out);
    else
        bev_mlp_kernel<false><<<NUM_ROWS / 16, 256, 0, stream>>>(ref, w1, b1, w2,
                                                                 w2t, b2, out);

    if (useNHWC)
        bev_samp_kernel<true><<<(NUM_ROWS + 255) / 256, 256, 0, stream>>>(
            f0, f1, f2, f3, ref, l2i, nhwc, out);
    else
        bev_samp_kernel<false><<<(NUM_ROWS + 255) / 256, 256, 0, stream>>>(
            f0, f1, f2, f3, ref, l2i, nhwc, out);
}
